// RWFLayer_35854386987282
// MI455X (gfx1250) — compile-verified, matrix operand order checked by reading, not run on hardware
//
#include <hip/hip_runtime.h>
#include <cmath>

typedef __attribute__((ext_vector_type(16))) _Float16 v16h;
typedef __attribute__((ext_vector_type(8)))  float    v8f;

#define RWF_BATCH 1048576
#define RWF_K     128
#define RWF_N     128
#define STRIPS    4            // 16-row strips per wave

// ---------------------------------------------------------------------------
// Kernel 1: W16[frag layout] = f16( exp(s[k]) * V[k][n] )
//
// B-fragment layout for V_WMMA_F32_16X16X32_F16 (wave32), per CDNA5 ISA:
//   fragment f = kc*8 + nt   (kc = K-chunk of 32, nt = N-tile of 16)
//   lane l: column n = nt*16 + (l & 15); hi = l >> 4
//   half  h (0..15): k = kc*32 + (h & 7) + hi*8 + ((h & 8) ? 16 : 0)
// Stored flat: wfrag[f*512 + l*16 + h], so each lane's 16 halves (32 B) are
// contiguous -> the GEMM kernel stages it with plain b128 copies.
// ---------------------------------------------------------------------------
__global__ void rwf_prep_wfrag(const float* __restrict__ s,
                               const float* __restrict__ V,
                               _Float16* __restrict__ wfrag) {
    int idx    = blockIdx.x * blockDim.x + threadIdx.x;   // 0 .. 16383
    int frag   = idx >> 9;                                 // 512 halves / frag
    int within = idx & 511;
    int lane   = within >> 4;
    int h      = within & 15;
    int kc = frag >> 3;
    int nt = frag & 7;
    int hi = lane >> 4;
    int ln = lane & 15;
    int k = kc * 32 + (h & 7) + hi * 8 + ((h & 8) ? 16 : 0);
    int n = nt * 16 + ln;
    float w = __expf(s[k]) * V[k * RWF_N + n];
    wfrag[idx] = (_Float16)w;
}

__device__ __forceinline__ float fast_tanh(float a) {
#if __has_builtin(__builtin_amdgcn_tanhf)
    return __builtin_amdgcn_tanhf(a);          // v_tanh_f32
#else
    float e = __expf(-2.0f * fabsf(a));        // v_exp_f32
    float t = (1.0f - e) * __builtin_amdgcn_rcpf(1.0f + e);  // v_rcp_f32
    return copysignf(t, a);
#endif
}

// ---------------------------------------------------------------------------
// Kernel 2: out = tanh(x @ W + bias); each wave owns STRIPS 16x128 strips.
// ---------------------------------------------------------------------------
__global__ __launch_bounds__(256)
void rwf_gemm_tanh(const float* __restrict__ x,
                   const _Float16* __restrict__ wfrag,
                   const float* __restrict__ bias,
                   float* __restrict__ out) {
    __shared__ __align__(32) _Float16 ldsW[32 * 512];     // 32 KB

    const int tid = threadIdx.x;

    // Stage pre-fragmentized W into LDS: 256 threads x 8 x 16B, coalesced.
    {
        const float4* src = (const float4*)wfrag;
        float4*       dst = (float4*)ldsW;
#pragma unroll
        for (int i = 0; i < 8; ++i)
            dst[tid + i * 256] = src[tid + i * 256];
    }
    __syncthreads();

    const int lane = tid & 31;
    const int wave = tid >> 5;
    const int hi   = lane >> 4;
    const int ln   = lane & 15;

    float bv[8];
#pragma unroll
    for (int nt = 0; nt < 8; ++nt) bv[nt] = bias[nt * 16 + ln];

    const long strip0 = ((long)blockIdx.x * 8 + wave) * STRIPS;

    for (int it = 0; it < STRIPS; ++it) {
        const long  m0 = (strip0 + it) * 16;              // first row of strip
        const float* xr = x + (m0 + ln) * RWF_K;          // this lane's A row

        if (it + 1 < STRIPS) {                            // warm L2 for next strip
            __builtin_prefetch(xr + 16 * RWF_K, 0, 2);
            __builtin_prefetch(xr + 16 * RWF_K + 64, 0, 2);
        }

        v8f acc[8] = {};

#pragma unroll
        for (int kc = 0; kc < 4; ++kc) {
            const int kb0 = kc * 32 + hi * 8;    // K run 1 (8 contiguous f32)
            const int kb1 = kb0 + 16;            // K run 2
            float4 a0 = *(const float4*)(xr + kb0);
            float4 a1 = *(const float4*)(xr + kb0 + 4);
            float4 a2 = *(const float4*)(xr + kb1);
            float4 a3 = *(const float4*)(xr + kb1 + 4);

            v16h af;                              // A fragment, f32 -> f16
            af[0]  = (_Float16)a0.x;  af[1]  = (_Float16)a0.y;
            af[2]  = (_Float16)a0.z;  af[3]  = (_Float16)a0.w;
            af[4]  = (_Float16)a1.x;  af[5]  = (_Float16)a1.y;
            af[6]  = (_Float16)a1.z;  af[7]  = (_Float16)a1.w;
            af[8]  = (_Float16)a2.x;  af[9]  = (_Float16)a2.y;
            af[10] = (_Float16)a2.z;  af[11] = (_Float16)a2.w;
            af[12] = (_Float16)a3.x;  af[13] = (_Float16)a3.y;
            af[14] = (_Float16)a3.z;  af[15] = (_Float16)a3.w;

            // Pull all 8 B-fragments of this K-chunk first (one DS clause),
            // then issue the 8 WMMAs back-to-back into the matrix pipe.
            v16h bf[8];
#pragma unroll
            for (int nt = 0; nt < 8; ++nt)
                bf[nt] = *(const v16h*)&ldsW[(kc * 8 + nt) * 512 + lane * 16];

#pragma unroll
            for (int nt = 0; nt < 8; ++nt)
                acc[nt] = __builtin_amdgcn_wmma_f32_16x16x32_f16(
                    false, af, false, bf[nt], (short)0, acc[nt], false, false);
        }

        // Epilogue: bias + tanh, store.
        // C/D layout: VGPR r of acc[nt] -> row m0 + hi*8 + r, col nt*16 + ln.
#pragma unroll
        for (int r = 0; r < 8; ++r) {
            const long orow = (m0 + hi * 8 + r) * (long)RWF_N;
#pragma unroll
            for (int nt = 0; nt < 8; ++nt) {
                float a = acc[nt][r] + bv[nt];
                out[orow + nt * 16 + ln] = fast_tanh(a);
            }
        }
    }
}

// ---------------------------------------------------------------------------
extern "C" void kernel_launch(void* const* d_in, const int* in_sizes, int n_in,
                              void* d_out, int out_size, void* d_ws, size_t ws_size,
                              hipStream_t stream) {
    const float* x    = (const float*)d_in[0];
    const float* s    = (const float*)d_in[1];
    const float* V    = (const float*)d_in[2];
    const float* bias = (const float*)d_in[3];
    float*       out  = (float*)d_out;
    _Float16*    wfrag = (_Float16*)d_ws;       // 32 KB of workspace

    // 128*128 = 16384 fragment elements
    rwf_prep_wfrag<<<64, 256, 0, stream>>>(s, V, wfrag);

    // 1048576 rows / (8 waves * 16 rows * STRIPS) = 2048 workgroups
    rwf_gemm_tanh<<<RWF_BATCH / (128 * STRIPS), 256, 0, stream>>>(x, wfrag, bias, out);
}
